// PatchMerging3D_78735340470550
// MI455X (gfx1250) — matrix operand order repaired and checked by reading, not compiled, not run on hardware
//
#include <hip/hip_runtime.h>

// ---- types for CDNA5 WMMA (gfx1250, wave32) ----
typedef _Float16 v8h  __attribute__((ext_vector_type(8)));
typedef _Float16 v16h __attribute__((ext_vector_type(16)));
typedef float    v8f  __attribute__((ext_vector_type(8)));

#define K_DIM 768      // 8*C
#define N_DIM 192      // 2*C
#define ROWS  32       // output rows per workgroup
#define LDA   776      // padded LDS row stride (halves): 776*2B = 1552B -> bank stride 4
#define LNEPS 1e-5f

// Problem constants (B=2, C=96, D=32, H=128, W=128)
#define XB_STRIDE 50331648L   // C*D*H*W
#define XC_STRIDE 524288L     // D*H*W
#define XD_STRIDE 16384      // H*W
#define XH_STRIDE 128        // W
#define SPATIAL   65536      // D2*H2*W2 = 16*64*64

__global__ __launch_bounds__(256)
void pm3d_wconv(const float* __restrict__ w, _Float16* __restrict__ wh) {
    int i = blockIdx.x * 256 + threadIdx.x;   // 192*768 = 147456 elements, grid sized exactly
    wh[i] = (_Float16)w[i];
}

__global__ __launch_bounds__(256)
void pm3d_main(const float* __restrict__ x,
               const float* __restrict__ gamma,
               const float* __restrict__ beta,
               const _Float16* __restrict__ wh,   // f16 weights, row-major [192][768]
               float* __restrict__ out) {
    __shared__ __align__(16) _Float16 As[ROWS * LDA];  // normalized f16 A-tile
    __shared__ float red[ROWS][8][2];                  // per-octant partial sums
    __shared__ float mv[ROWS][2];                      // mean, rsigma per row

    const int t  = threadIdx.x;
    const int wg = blockIdx.x;

    // ================= Phase 1: gather + LayerNorm stats =================
    {
        const int row = t >> 3;            // 0..31 local row
        const int blk = t & 7;             // octant (dd,hh,ww)
        const int dd = blk >> 2, hh = (blk >> 1) & 1, ww = blk & 1;
        const int r  = wg * ROWS + row;
        const int w2 = r & 63, h2 = (r >> 6) & 63, d2 = (r >> 12) & 15, b = r >> 16;
        const float* xp = x + (long)b * XB_STRIDE
                            + (long)(2 * d2 + dd) * XD_STRIDE
                            + (2 * h2 + hh) * XH_STRIDE
                            + (2 * w2 + ww);
        _Float16* ap = As + row * LDA + blk * 96;   // e = 384*dd + 192*hh + 96*ww + c
        float s = 0.f, q = 0.f;
        #pragma unroll 4
        for (int c = 0; c < 96; ++c) {
            float v = xp[(long)c * XC_STRIDE];
            s += v; q += v * v;
            ap[c] = (_Float16)v;                    // stash raw for pass 2
        }
        red[row][blk][0] = s;
        red[row][blk][1] = q;
    }
    __syncthreads();
    if (t < ROWS) {
        float s = 0.f, q = 0.f;
        #pragma unroll
        for (int j = 0; j < 8; ++j) { s += red[t][j][0]; q += red[t][j][1]; }
        float mean = s * (1.0f / 768.0f);
        float var  = q * (1.0f / 768.0f) - mean * mean;
        mv[t][0] = mean;
        mv[t][1] = __frsqrt_rn(var + LNEPS);
    }
    __syncthreads();
    {
        const int row = t >> 3, blk = t & 7;
        const float mean = mv[row][0], rs = mv[row][1];
        _Float16* ap = As + row * LDA + blk * 96;
        const float* gp = gamma + blk * 96;
        const float* bp = beta  + blk * 96;
        #pragma unroll 4
        for (int c = 0; c < 96; ++c) {
            float v = (float)ap[c];
            ap[c] = (_Float16)((v - mean) * rs * gp[c] + bp[c]);
        }
    }
    __syncthreads();

    // ================= Phase 2: WMMA GEMM (32 x 768) @ (768 x 192) =================
    const int wv   = t >> 5;        // wave 0..7
    const int lane = t & 31;
    const int lrow = lane & 15;
    const int hi   = lane >> 4;     // 0: lanes 0-15, 1: lanes 16-31
    const int r0   = wg * ROWS;
    const int b    = r0 >> 16;

    // 24 wave-tiles (2 row-tiles x 12 col-tiles), 3 per wave
    #pragma unroll
    for (int it = 0; it < 3; ++it) {
        const int tt = wv + 8 * it;
        const int rt = tt & 1;          // row tile 0/1
        const int nt = tt >> 1;         // col tile 0..11
        const int n  = nt * 16 + lrow;  // output channel this lane owns (A/B striping)

        // A: lane holds row (rt*16+lrow), K contiguous hi*16..hi*16+15 (matches B)
        const _Float16* aptr = As + (rt * 16 + lrow) * LDA + hi * 16;
        // B: lane holds column n; lanes 0-15 carry K k0..k0+15, lanes 16-31 carry k0+16..k0+31
        const _Float16* bptr = wh + n * K_DIM + hi * 16;

        v8f acc = {};
        #pragma unroll
        for (int kk = 0; kk < 24; ++kk) {
            const int k0 = kk * 32;
            v8h alo = *(const v8h*)(aptr + k0);         // ds_load_b128
            v8h ahi = *(const v8h*)(aptr + k0 + 8);     // ds_load_b128
            v16h a;
            #pragma unroll
            for (int j = 0; j < 8; ++j) { a[j] = alo[j]; a[j + 8] = ahi[j]; }
            v16h bm = *(const v16h*)(bptr + k0);        // 32B contiguous global load (L2-hot)
            acc = __builtin_amdgcn_wmma_f32_16x16x32_f16(
                    /*neg_a=*/false, a, /*neg_b=*/false, bm,
                    /*c_mod=*/(short)0, acc, /*reuse_a=*/false, /*reuse_b=*/false);
        }

        // C layout: VGPR i -> M = i + hi*8 (consecutive spatial idx), N = lrow
        const int sbase = (r0 - (b << 16)) + rt * 16 + hi * 8;  // spatial index of acc[0]
        float* op = out + (long)(b * N_DIM + n) * SPATIAL + sbase;
        *(float4*)op       = make_float4(acc[0], acc[1], acc[2], acc[3]);
        *((float4*)op + 1) = make_float4(acc[4], acc[5], acc[6], acc[7]);
    }
}

extern "C" void kernel_launch(void* const* d_in, const int* in_sizes, int n_in,
                              void* d_out, int out_size, void* d_ws, size_t ws_size,
                              hipStream_t stream) {
    const float* x     = (const float*)d_in[0];
    const float* w     = (const float*)d_in[1];
    const float* gamma = (const float*)d_in[2];
    const float* beta  = (const float*)d_in[3];
    float* out = (float*)d_out;
    _Float16* wh = (_Float16*)d_ws;   // 192*768*2 = 294912 bytes of scratch

    // 1) convert weights to f16 once per launch (deterministic, capture-safe)
    pm3d_wconv<<<(N_DIM * K_DIM) / 256, 256, 0, stream>>>(w, wh);

    // 2) fused gather + LayerNorm + WMMA GEMM: 131072 rows / 32 = 4096 WGs
    pm3d_main<<<4096, 256, 0, stream>>>(x, gamma, beta, wh, out);
}
